// KSSMBlock_66872640798748
// MI455X (gfx1250) — matrix operand order repaired and checked by reading, not compiled, not run on hardware
//
#include <hip/hip_runtime.h>
#include <hip/hip_bf16.h>

// ---------------- problem constants ----------------
#define BSZ     2
#define LSEQ    2048
#define DM      1024          // D_MODEL
#define DI      2048          // D_INNER
#define DCONV   4
#define RANK    64            // DT_RANK
#define DSTATE  16
#define NPROJ   (2*DI + 2*DSTATE + RANK)   // 4192
#define MTOK    (BSZ*LSEQ)                 // 4096

typedef __attribute__((ext_vector_type(16))) _Float16 v16h;
typedef __attribute__((ext_vector_type(8)))  _Float16 v8h;
typedef __attribute__((ext_vector_type(8)))  float    v8f;

__device__ __forceinline__ float softplusf(float v) {
    return (v > 20.f) ? v : logf(1.f + __expf(v));
}
__device__ __forceinline__ float siluf(float v) {
    return v / (1.f + __expf(-v));
}

// ---------------- f32 -> f16 convert ----------------
__global__ void k_cvt_f16(const float* __restrict__ src, _Float16* __restrict__ dst, int n) {
    int i = blockIdx.x * blockDim.x + threadIdx.x;
    if (i < n) dst[i] = (_Float16)src[i];
}

// ---------------- LayerNorm (one block per token) ----------------
__global__ void k_layernorm(const float* __restrict__ x, const float* __restrict__ w,
                            const float* __restrict__ b, _Float16* __restrict__ xn) {
    __shared__ float s1[256];
    __shared__ float s2[256];
    const int row = blockIdx.x;
    const float* xr = x + (size_t)row * DM;
    float s = 0.f, sq = 0.f;
    for (int i = threadIdx.x; i < DM; i += 256) { float v = xr[i]; s += v; sq += v * v; }
    s1[threadIdx.x] = s; s2[threadIdx.x] = sq;
    __syncthreads();
    for (int off = 128; off > 0; off >>= 1) {
        if (threadIdx.x < off) { s1[threadIdx.x] += s1[threadIdx.x + off];
                                 s2[threadIdx.x] += s2[threadIdx.x + off]; }
        __syncthreads();
    }
    const float mean = s1[0] * (1.f / DM);
    const float var  = s2[0] * (1.f / DM) - mean * mean;
    const float rstd = rsqrtf(var + 1e-5f);
    for (int i = threadIdx.x; i < DM; i += 256)
        xn[(size_t)row * DM + i] = (_Float16)((xr[i] - mean) * rstd * w[i] + b[i]);
}

// ---------------- WMMA GEMM helpers ----------------
__device__ __forceinline__ void load_frags(const _Float16* const arow[4],
                                           const _Float16* const brow[4],
                                           int kk, int kA, int kB,
                                           v16h afr[4], v16h bfr[4]) {
#pragma unroll
    for (int i = 0; i < 4; ++i) {
        v8h alo = *(const v8h*)(arow[i] + kk + kA);
        v8h ahi = *(const v8h*)(arow[i] + kk + kA + 16);
#pragma unroll
        for (int e = 0; e < 8; ++e) { afr[i][e] = alo[e]; afr[i][8 + e] = ahi[e]; }
        v8h blo = *(const v8h*)(brow[i] + kk + kB);
        v8h bhi = *(const v8h*)(brow[i] + kk + kB + 16);
#pragma unroll
        for (int e = 0; e < 8; ++e) { bfr[i][e] = blo[e]; bfr[i][8 + e] = bhi[e]; }
    }
}

__device__ __forceinline__ void mma_4x4(const v16h afr[4], const v16h bfr[4],
                                        v8f acc[4][4]) {
#pragma unroll
    for (int i = 0; i < 4; ++i)
#pragma unroll
        for (int j = 0; j < 4; ++j)
            acc[i][j] = __builtin_amdgcn_wmma_f32_16x16x32_f16(
                false, afr[i], false, bfr[j], (short)0, acc[i][j], false, false);
}

// ---------------- WMMA GEMM: C[M,N] (+res) = A[M,K] * B[N,K]^T ----------------
// Register-blocked 64x64 tile per wave (4x4 fragments, 16 WMMAs / 32-wide K step)
// with a software-pipelined, double-buffered K loop (K % 64 == 0 for both GEMMs).
__global__ void __launch_bounds__(256)
k_gemm_wmma(const _Float16* __restrict__ A,
            const _Float16* __restrict__ Bw,
            const float* __restrict__ residual,
            float* __restrict__ C,
            int M, int N, int K) {
    const int lane = threadIdx.x & 31;
    const int wave = threadIdx.x >> 5;
    const int m0 = blockIdx.y * 64;
    const int n0 = (blockIdx.x * 8 + wave) * 64;
    if (n0 >= N) return;                       // uniform per wave: EXEC stays all-ones

    const int rl = lane & 15;
    const int kA = (lane & 16) ? 8 : 0;        // A frag: K chunks {kA..kA+7, kA+16..kA+23}
    const int kB = (lane & 16) ? 8 : 0;        // B frag: K chunks {kB..kB+7, kB+16..kB+23}

    const _Float16* arow[4];
    const _Float16* brow[4];
    bool nok[4];
#pragma unroll
    for (int i = 0; i < 4; ++i) {
        arow[i] = A + (size_t)(m0 + i * 16 + rl) * K;
        const int nc = n0 + i * 16;
        nok[i] = (nc < N);                     // N is a multiple of 16
        brow[i] = Bw + (size_t)((nok[i] ? nc : 0) + rl) * K;   // clamp keeps loads in-bounds
    }

    v8f acc[4][4] = {};
    v16h a0[4], b0[4], a1[4], b1[4];
    load_frags(arow, brow, 0, kA, kB, a0, b0);
    for (int kk = 0; kk < K; kk += 64) {
        if ((kk & 255) == 0 && kk + 256 < K) {
#pragma unroll
            for (int i = 0; i < 4; ++i) {
                __builtin_prefetch(arow[i] + kk + 256, 0, 3);
                __builtin_prefetch(brow[i] + kk + 256, 0, 3);
            }
        }
        load_frags(arow, brow, kk + 32, kA, kB, a1, b1);   // fetch next half-step
        mma_4x4(a0, b0, acc);                              // compute current
        if (kk + 64 < K)
            load_frags(arow, brow, kk + 64, kA, kB, a0, b0);
        mma_4x4(a1, b1, acc);
    }

    const int mb = (lane & 16) ? 8 : 0;
#pragma unroll
    for (int i = 0; i < 4; ++i) {
#pragma unroll
        for (int j = 0; j < 4; ++j) {
            if (!nok[j]) continue;
            const int colj = n0 + j * 16 + rl;
#pragma unroll
            for (int r = 0; r < 8; ++r) {
                size_t o = (size_t)(m0 + i * 16 + mb + r) * N + colj;
                float v = acc[i][j][r];
                if (residual) v += residual[o];
                C[o] = v;
            }
        }
    }
}

// ---------------- depthwise causal conv (taps=4) + SiLU ----------------
__global__ void k_conv_silu(const float* __restrict__ proj,
                            const float* __restrict__ cw, const float* __restrict__ cb,
                            float* __restrict__ xact) {
    int idx = blockIdx.x * blockDim.x + threadIdx.x;     // token*DI + ch
    if (idx >= MTOK * DI) return;
    int tok = idx / DI, ch = idx % DI;
    int b = tok / LSEQ, t = tok % LSEQ;
    float acc = cb[ch];
#pragma unroll
    for (int j = 0; j < DCONV; ++j) {
        int tt = t - (DCONV - 1) + j;
        float v = (tt >= 0) ? proj[(size_t)(b * LSEQ + tt) * NPROJ + DI + ch] : 0.f;
        acc += cw[ch * DCONV + j] * v;
    }
    xact[idx] = siluf(acc);
}

// ---------------- rank-64 projections -> alpha / omega / dt ----------------
__global__ void k_dtproj(const float* __restrict__ proj,
                         const float* __restrict__ aw, const float* __restrict__ ab,
                         const float* __restrict__ ow, const float* __restrict__ ob,
                         const float* __restrict__ dw, const float* __restrict__ db,
                         float* __restrict__ alpha, float* __restrict__ omega,
                         float* __restrict__ dt) {
    __shared__ float dtp[RANK];
    const int tok = blockIdx.x;
    if (threadIdx.x < RANK)
        dtp[threadIdx.x] = proj[(size_t)tok * NPROJ + 2 * DI + 2 * DSTATE + threadIdx.x];
    __syncthreads();
    const int ch = blockIdx.y * 256 + threadIdx.x;
    float sa = 0.f, so = 0.f, sd = 0.f;
#pragma unroll 8
    for (int k = 0; k < RANK; ++k) {
        float d = dtp[k];
        sa += d * aw[ch * RANK + k];
        so += d * ow[ch * RANK + k];
        sd += d * dw[ch * RANK + k];
    }
    size_t idx = (size_t)tok * DI + ch;
    alpha[idx] = softplusf(sa + ab[ch]);
    omega[idx] = so + ob[ch];
    dt[idx]    = softplusf(sd + db[ch]);
}

// ---------------- sequential rotation-SSM scan (one thread per (b,ch) chain) ----
__global__ void k_scan(const float* __restrict__ proj, const float* __restrict__ xact,
                       const float* __restrict__ alpha, const float* __restrict__ omega,
                       const float* __restrict__ dtb,
                       const float* __restrict__ state0,
                       const float* __restrict__ Cw, const float* __restrict__ Dp,
                       _Float16* __restrict__ yh, float* __restrict__ state_out) {
    int g = blockIdx.x * blockDim.x + threadIdx.x;       // 0 .. B*DI-1
    int b = g / DI, ch = g % DI;
    int grp = ch / (DI / DSTATE);                        // 128 channels per state group
    float h0 = state0[(size_t)g * 2 + 0];
    float h1 = state0[(size_t)g * 2 + 1];
    const float cw0 = Cw[ch * 2], cw1 = Cw[ch * 2 + 1], dpar = Dp[ch];
    for (int t = 0; t < LSEQ; ++t) {
        int tok = b * LSEQ + t;
        size_t idx = (size_t)tok * DI + ch;
        const float* pr = proj + (size_t)tok * NPROJ;
        float dt = dtb[idx], al = alpha[idx], om = omega[idx], xa = xact[idx];
        float B0 = pr[2 * DI + grp * 2], B1 = pr[2 * DI + grp * 2 + 1];
        float z  = pr[ch];
        float tau  = 0.5f * dt;
        float opta = 1.f + tau * al;
        float tw   = tau * om;
        float inv  = 1.f / (opta * opta + tw * tw + 1e-6f);
        float m11 = opta * inv, m12 = tw * inv;
        float omta = 1.f - tau * al;
        float a11 =  m11 * omta - m12 * tw;
        float a12 =  m11 * tw  + m12 * omta;
        float a21 = -m12 * omta - m11 * tw;
        float a22 = -m12 * tw  + m11 * omta;
        float Bx0 = B0 * xa, Bx1 = B1 * xa;
        float u0 = dt * ( m11 * Bx0 + m12 * Bx1);
        float u1 = dt * (-m12 * Bx0 + m11 * Bx1);
        float nh0 = a11 * h0 + a12 * h1 + u0;
        float nh1 = a21 * h0 + a22 * h1 + u1;
        h0 = nh0; h1 = nh1;
        float y = h0 * cw0 + h1 * cw1 + dpar * xa;
        yh[idx] = (_Float16)(y * siluf(z));
    }
    state_out[(size_t)g * 2 + 0] = h0;
    state_out[(size_t)g * 2 + 1] = h1;
}

// ---------------- launch ----------------
extern "C" void kernel_launch(void* const* d_in, const int* in_sizes, int n_in,
                              void* d_out, int out_size, void* d_ws, size_t ws_size,
                              hipStream_t stream) {
    const float* x       = (const float*)d_in[0];
    const float* state   = (const float*)d_in[1];
    const float* norm_w  = (const float*)d_in[2];
    const float* norm_b  = (const float*)d_in[3];
    const float* inpw    = (const float*)d_in[4];
    const float* conv_w  = (const float*)d_in[5];
    const float* conv_b  = (const float*)d_in[6];
    const float* alpha_w = (const float*)d_in[7];
    const float* alpha_b = (const float*)d_in[8];
    const float* omega_w = (const float*)d_in[9];
    const float* omega_b = (const float*)d_in[10];
    const float* dt_w    = (const float*)d_in[11];
    const float* dt_b    = (const float*)d_in[12];
    const float* C_w     = (const float*)d_in[13];
    const float* D_p     = (const float*)d_in[14];
    const float* out_w   = (const float*)d_in[15];

    float* out_main  = (float*)d_out;                      // (B,L,DM)
    float* out_state = (float*)d_out + (size_t)MTOK * DM;  // (B,DI,2)

    char* p = (char*)d_ws;
    _Float16* xn    = (_Float16*)p; p += (size_t)MTOK  * DM * 2;
    _Float16* winh  = (_Float16*)p; p += (size_t)NPROJ * DM * 2;
    float*    proj  = (float*)p;    p += (size_t)MTOK  * NPROJ * 4;
    float*    xact  = (float*)p;    p += (size_t)MTOK  * DI * 4;
    float*    alpha = (float*)p;    p += (size_t)MTOK  * DI * 4;
    float*    omega = (float*)p;    p += (size_t)MTOK  * DI * 4;
    float*    dtbuf = (float*)p;    p += (size_t)MTOK  * DI * 4;
    _Float16* yh    = (_Float16*)p; p += (size_t)MTOK  * DI * 2;
    _Float16* wouth = (_Float16*)p; p += (size_t)DM    * DI * 2;

    // weight conversions (pure function of inputs -> safe to redo every call)
    {
        int n = NPROJ * DM;
        k_cvt_f16<<<(n + 255) / 256, 256, 0, stream>>>(inpw, winh, n);
        n = DM * DI;
        k_cvt_f16<<<(n + 255) / 256, 256, 0, stream>>>(out_w, wouth, n);
    }
    // 1) layernorm -> f16 activations
    k_layernorm<<<MTOK, 256, 0, stream>>>(x, norm_w, norm_b, xn);
    // 2) in_proj GEMM  (M=4096, N=4192, K=1024): block tile 64x512, wave tile 64x64
    k_gemm_wmma<<<dim3((NPROJ + 511) / 512, MTOK / 64), 256, 0, stream>>>(
        xn, winh, nullptr, proj, MTOK, NPROJ, DM);
    // 3) depthwise conv + SiLU
    k_conv_silu<<<(MTOK * DI + 255) / 256, 256, 0, stream>>>(proj, conv_w, conv_b, xact);
    // 4) alpha / omega / dt rank-64 projections
    k_dtproj<<<dim3(MTOK, DI / 256), 256, 0, stream>>>(
        proj, alpha_w, alpha_b, omega_w, omega_b, dt_w, dt_b, alpha, omega, dtbuf);
    // 5) sequential scan + gating -> y (f16) and final state
    k_scan<<<(BSZ * DI) / 256, 256, 0, stream>>>(
        proj, xact, alpha, omega, dtbuf, state, C_w, D_p, yh, out_state);
    // 6) out GEMM + residual  (M=4096, N=1024, K=2048)
    k_gemm_wmma<<<dim3((DM + 511) / 512, MTOK / 64), 256, 0, stream>>>(
        yh, wouth, x, out_main, MTOK, DM, DI);
}
